// E2ESWIGLULayer_44367012168399
// MI455X (gfx1250) — matrix operand order repaired and checked by reading, not compiled, not run on hardware
//
#include <hip/hip_runtime.h>
#include <hip/hip_bf16.h>

// ---------------- problem constants (match reference) ----------------
constexpr int B_   = 2;
constexpr int S_   = 2048;
constexpr int H_   = 2048;
constexpr int NH_  = 16;
constexpr int HD_  = 128;           // H / NH
constexpr int WIN_ = 1024;

typedef __attribute__((ext_vector_type(16))) __bf16 v16bf;
typedef __attribute__((ext_vector_type(8)))  float  v8f;
typedef __attribute__((ext_vector_type(4)))  unsigned int v4u;
typedef __attribute__((ext_vector_type(8)))  int v8i;
typedef __attribute__((ext_vector_type(4)))  int v4i;

union FragB { v16bf v; unsigned u[8]; };

__device__ __forceinline__ unsigned short f2bf(float f) {
  unsigned x = __float_as_uint(f);
  unsigned r = x + 0x7fffu + ((x >> 16) & 1u);   // round-to-nearest-even
  return (unsigned short)(r >> 16);
}

// ---------------- Tensor Data Mover: 2D bf16 tile Global -> LDS ----------------
// D# per CDNA5 ISA ch.8: group0 = {count/lds_addr/global_addr/type}, group1 =
// {data_size, tensor dims, tile dims, dim0 stride}. 2D tile => groups 2/3 zero.
// This toolchain exposes the 6-arg builtin (extra int32x8 group before cpol).
__device__ __forceinline__ void tdm_load_2d_bf16(const void* gaddr, unsigned lds_off,
                                                 int tensor_w, int tensor_h,
                                                 int stride_elems,
                                                 int tile_w, int tile_h) {
  unsigned long long ga = (unsigned long long)(size_t)gaddr;
  v4u g0;
  g0[0] = 1u;                                            // count=1 (valid), user mode
  g0[1] = lds_off;                                       // LDS byte address
  g0[2] = (unsigned)(ga & 0xffffffffu);                  // global_addr[31:0]
  g0[3] = (unsigned)((ga >> 32) & 0x1ffffffu) | (2u << 30); // addr[56:32] | type=2
  v8i g1;
  g1[0] = (int)(1u << 16);                               // data_size=1 -> 2 bytes
  g1[1] = (int)(((unsigned)tensor_w & 0xffffu) << 16);   // tensor_dim0[15:0]
  g1[2] = (int)((((unsigned)tensor_w >> 16) & 0xffffu) |
                (((unsigned)tensor_h & 0xffffu) << 16)); // dim0[31:16] | dim1[15:0]
  g1[3] = (int)((((unsigned)tensor_h >> 16) & 0xffffu) |
                (((unsigned)tile_w & 0xffffu) << 16));   // dim1[31:16] | tile_dim0
  g1[4] = (int)((unsigned)tile_h & 0xffffu);             // tile_dim1 | tile_dim2=0
  g1[5] = stride_elems;                                  // tensor_dim0_stride[31:0]
  g1[6] = 0;
  g1[7] = 0;
  v4i z4 = {0, 0, 0, 0};
  v8i z8 = {0, 0, 0, 0, 0, 0, 0, 0};
  __builtin_amdgcn_tensor_load_to_lds(g0, g1, z4, z4, z8, 0);
}

// ---------------- fp32 -> bf16 conversion ----------------
__global__ void cvt_f32_bf16(const float* __restrict__ in,
                             unsigned short* __restrict__ out, size_t n) {
  size_t i = (size_t)blockIdx.x * blockDim.x + threadIdx.x;
  if (i < n) out[i] = f2bf(in[i]);
}

// ---------------- bf16 WMMA GEMM: C(f32, MxN) = A(MxK) * B(KxN) ----------------
// block tile 128x128, K-step 32, 256 threads = 8 waves (4 along M x 2 along N).
// A tile staged via Tensor Data Mover (async, TENSORcnt); B staged transposed
// by the workgroup. N,K compile-time so C-store offsets fold to immediates.
template <int N, int K>
__global__ __launch_bounds__(256)
void gemm_bf16(const unsigned short* __restrict__ A,
               const unsigned short* __restrict__ B,
               float* __restrict__ C) {
  __shared__ unsigned short sA[128 * 32];   // [m][k]   (TDM destination)
  __shared__ unsigned short sB[128 * 32];   // [n][k]   (transposed stage)

  const int tid  = threadIdx.x;
  const int lane = tid & 31;
  const int wave = tid >> 5;
  const int wm   = (wave & 3) * 32;     // wave M offset within block
  const int wn   = (wave >> 2) * 64;    // wave N offset within block
  const int m0   = blockIdx.y * 128;
  const int n0   = blockIdx.x * 128;
  const int hi   = lane >> 4;           // lane half
  const int l15  = lane & 15;

  const unsigned ldsA = (unsigned)(size_t)(void*)sA;   // low 32b of flat = LDS offset

  const v8f vzero = {0.f,0.f,0.f,0.f,0.f,0.f,0.f,0.f};
  v8f acc[2][4];
  #pragma unroll
  for (int i = 0; i < 2; i++)
    #pragma unroll
    for (int j = 0; j < 4; j++) acc[i][j] = vzero;

  for (int k0 = 0; k0 < K; k0 += 32) {
    // --- A tile 128x32 via TDM (row-major, contiguous rows in LDS) ---
    if (wave == 0) {
      tdm_load_2d_bf16(A + (size_t)m0 * K + k0, ldsA,
                       /*tensor_w=*/K, /*tensor_h=*/128 * (int)gridDim.y,
                       /*stride=*/K, /*tile_w=*/32, /*tile_h=*/128);
    }
    // --- B tile staged transposed: coalesced global rows -> [n][k] in LDS ---
    {
      int k = tid >> 3, ng = tid & 7;
      const uint4* src = (const uint4*)(B + (size_t)(k0 + k) * N + n0 + ng * 16);
      if (k0 + 32 < K)
        __builtin_prefetch((const void*)(B + (size_t)(k0 + 32 + k) * N + n0 + ng * 16), 0, 1);
      union { uint4 q[2]; unsigned short s[16]; } t;
      t.q[0] = src[0]; t.q[1] = src[1];
      int nb = ng * 16;
      #pragma unroll
      for (int i = 0; i < 16; i++) sB[(nb + i) * 32 + k] = t.s[i];
    }
    if (wave == 0) __builtin_amdgcn_s_wait_tensorcnt(0);
    __syncthreads();

    FragB af[2], bf[4];
    { // A fragments: row = lane, K 0..15 / 16..31 by lane half
      int kbase = hi * 16;
      #pragma unroll
      for (int mt = 0; mt < 2; mt++) {
        const unsigned* rp = (const unsigned*)(sA + (wm + mt * 16 + l15) * 32);
        #pragma unroll
        for (int p = 0; p < 8; p++) {
          af[mt].u[p] = rp[(kbase >> 1) + p];
        }
      }
    }
    { // B fragments: column = lane, K 0..15 / 16..31 by lane half
      int kbase = hi * 16;
      #pragma unroll
      for (int nt = 0; nt < 4; nt++) {
        const unsigned* cp = (const unsigned*)(sB + (wn + nt * 16 + l15) * 32);
        #pragma unroll
        for (int p = 0; p < 8; p++) bf[nt].u[p] = cp[(kbase >> 1) + p];
      }
    }
    #pragma unroll
    for (int mt = 0; mt < 2; mt++)
      #pragma unroll
      for (int nt = 0; nt < 4; nt++)
        acc[mt][nt] = __builtin_amdgcn_wmma_f32_16x16x32_bf16(
            false, af[mt].v, false, bf[nt].v, (short)0, acc[mt][nt], false, false);
    __syncthreads();
  }

  // C layout: VGPR r, lane half hi -> row M = r + 8*hi, col N = lane&15.
  // N is compile-time: r*N folds into immediate store offsets.
  #pragma unroll
  for (int mt = 0; mt < 2; mt++)
    #pragma unroll
    for (int nt = 0; nt < 4; nt++) {
      float* cp = C + (size_t)(m0 + wm + mt * 16 + hi * 8) * N
                    + (n0 + wn + nt * 16 + l15);
      #pragma unroll
      for (int r = 0; r < 8; r++) cp[(size_t)r * N] = acc[mt][nt][r];
    }
}

// ---------------- fused RMSNorm (over full H) + RoPE + layout split ----------------
__global__ __launch_bounds__(256)
void norm_rope_split(const float* __restrict__ qkv,
                     const float* __restrict__ qw, const float* __restrict__ kw,
                     unsigned short* __restrict__ qb,
                     unsigned short* __restrict__ kb,
                     unsigned short* __restrict__ vT) {
  const int token = blockIdx.x;
  const int b = token / S_;
  const int s = token - b * S_;
  const float* qrow = qkv + (size_t)token * (3 * H_);
  const float* krow = qrow + H_;
  const float* vrow = qrow + 2 * H_;

  __shared__ float red[256];
  __shared__ float rs[2];
  const int tid = threadIdx.x;

  float a = 0.f;
  for (int i = tid; i < H_; i += 256) { float x = qrow[i]; a += x * x; }
  red[tid] = a; __syncthreads();
  for (int w = 128; w > 0; w >>= 1) { if (tid < w) red[tid] += red[tid + w]; __syncthreads(); }
  if (tid == 0) rs[0] = rsqrtf(red[0] / (float)H_ + 1e-6f);
  __syncthreads();
  a = 0.f;
  for (int i = tid; i < H_; i += 256) { float x = krow[i]; a += x * x; }
  red[tid] = a; __syncthreads();
  for (int w = 128; w > 0; w >>= 1) { if (tid < w) red[tid] += red[tid + w]; __syncthreads(); }
  if (tid == 0) rs[1] = rsqrtf(red[0] / (float)H_ + 1e-6f);
  __syncthreads();

  const float rq = rs[0], rk = rs[1];
  const float kfreq = 0.14391156f;              // ln(10000)/64
  for (int idx = tid; idx < NH_ * 64; idx += 256) {
    int head = idx >> 6, j = idx & 63;
    float ang = (float)s * __expf(-(float)j * kfreq);
    float sn, cs; __sincosf(ang, &sn, &cs);
    int i1 = head * HD_ + j, i2 = i1 + 64;
    float q1 = qrow[i1] * rq * qw[i1], q2 = qrow[i2] * rq * qw[i2];
    float k1 = krow[i1] * rk * kw[i1], k2 = krow[i2] * rk * kw[i2];
    size_t ob = (((size_t)b * NH_ + head) * S_ + s) * HD_;
    qb[ob + j]      = f2bf(q1 * cs - q2 * sn);
    qb[ob + j + 64] = f2bf(q2 * cs + q1 * sn);
    kb[ob + j]      = f2bf(k1 * cs - k2 * sn);
    kb[ob + j + 64] = f2bf(k2 * cs + k1 * sn);
  }
  for (int idx = tid; idx < H_; idx += 256) {
    int head = idx >> 7, hd = idx & 127;
    vT[(((size_t)b * NH_ + head) * HD_ + hd) * S_ + s] = f2bf(vrow[idx]);
  }
}

// ---------------- windowed flash attention, one wave per 16-query tile ----------------
__global__ __launch_bounds__(32)
void attn_wmma(const unsigned short* __restrict__ qb,
               const unsigned short* __restrict__ kb,
               const unsigned short* __restrict__ vT,
               unsigned short* __restrict__ ob) {
  const int nqt = S_ / 16;
  const int blk = blockIdx.x;
  const int q0  = (blk % nqt) * 16;
  const int h   = (blk / nqt) % NH_;
  const int b   =  blk / (nqt * NH_);
  const int lane = threadIdx.x;
  const int hi = lane >> 4, l15 = lane & 15;

  __shared__ float          sc[16 * 32];   // raw scores tile
  __shared__ unsigned short sp[16 * 32];   // softmax probs (bf16)
  __shared__ float          rowf[16];      // per-row rescale / inv-sum broadcast

  const unsigned short* qh = qb + (((size_t)b * NH_ + h) * S_) * HD_;
  const unsigned short* kh = kb + (((size_t)b * NH_ + h) * S_) * HD_;
  const unsigned short* vh = vT + (((size_t)b * NH_ + h) * HD_) * S_;

  FragB qf[4];
  {
    const unsigned* qp = (const unsigned*)(qh + (size_t)(q0 + l15) * HD_);
    #pragma unroll
    for (int c = 0; c < 4; c++) {
      int kb0 = c * 32 + hi * 16;
      #pragma unroll
      for (int p = 0; p < 8; p++) {
        qf[c].u[p] = qp[(kb0 >> 1) + p];
      }
    }
  }

  const v8f vzero = {0.f,0.f,0.f,0.f,0.f,0.f,0.f,0.f};
  v8f o[8];
  #pragma unroll
  for (int n = 0; n < 8; n++) o[n] = vzero;

  float m_i = -1e30f, l_i = 0.f;
  const float scale = 0.08838834764831845f;   // 1/sqrt(128)

  int ks = q0 - (WIN_ - 1); if (ks < 0) ks = 0; ks &= ~31;

  for (int kblk = ks; kblk <= q0 + 15; kblk += 32) {
    // --- scores: two 16x16 tiles, K-dim 128 = 4 WMMA chunks each ---
    #pragma unroll
    for (int t = 0; t < 2; t++) {
      v8f cacc = vzero;
      #pragma unroll
      for (int c = 0; c < 4; c++) {
        FragB kf;
        const unsigned* kp = (const unsigned*)(
            kh + (size_t)(kblk + t * 16 + l15) * HD_ + c * 32 + hi * 16);
        #pragma unroll
        for (int p = 0; p < 8; p++) kf.u[p] = kp[p];
        cacc = __builtin_amdgcn_wmma_f32_16x16x32_bf16(
            false, qf[c].v, false, kf.v, (short)0, cacc, false, false);
      }
      #pragma unroll
      for (int r = 0; r < 8; r++)
        sc[(r + hi * 8) * 32 + t * 16 + l15] = cacc[r];
    }
    __syncthreads();

    // --- online softmax row update: lanes 0..15 each own one query row ---
    if (lane < 16) {
      int qi = q0 + lane;
      float pv[32];
      float bm = -1e30f;
      #pragma unroll
      for (int c = 0; c < 32; c++) {
        int key = kblk + c;
        float v = sc[lane * 32 + c] * scale;
        bool ok = (key <= qi) && ((qi - key) < WIN_);
        v = ok ? v : -1e30f;
        pv[c] = v;
        bm = fmaxf(bm, v);
      }
      float nm = fmaxf(m_i, bm);
      float f  = __expf(m_i - nm);
      float sm = 0.f;
      #pragma unroll
      for (int c = 0; c < 32; c++) {
        float e = (pv[c] > -1e29f) ? __expf(pv[c] - nm) : 0.f;
        sp[lane * 32 + c] = f2bf(e);
        sm += e;
      }
      l_i = l_i * f + sm;
      m_i = nm;
      rowf[lane] = f;
    }
    __syncthreads();

    // --- rescale O accumulators (row M = r + 8*hi) ---
    float fr[8];
    #pragma unroll
    for (int r = 0; r < 8; r++) fr[r] = rowf[r + hi * 8];
    #pragma unroll
    for (int n = 0; n < 8; n++)
      #pragma unroll
      for (int r = 0; r < 8; r++) o[n][r] *= fr[r];

    // --- P fragment from LDS (A layout 16x32) ---
    FragB pf;
    {
      const unsigned* pp = (const unsigned*)(sp + l15 * 32);
      int kb0 = hi * 16;
      #pragma unroll
      for (int p = 0; p < 8; p++) {
        pf.u[p] = pp[(kb0 >> 1) + p];
      }
    }
    // --- P·V: V fragments from transposed V, contiguous key pairs ---
    #pragma unroll
    for (int n = 0; n < 8; n++) {
      FragB vf;
      const unsigned* vp = (const unsigned*)(
          vh + (size_t)(n * 16 + l15) * S_ + kblk + hi * 16);
      #pragma unroll
      for (int p = 0; p < 8; p++) vf.u[p] = vp[p];
      o[n] = __builtin_amdgcn_wmma_f32_16x16x32_bf16(
          false, pf.v, false, vf.v, (short)0, o[n], false, false);
    }
    __syncthreads();
  }

  if (lane < 16) rowf[lane] = (l_i > 0.f) ? (1.f / l_i) : 0.f;
  __syncthreads();
  float fr[8];
  #pragma unroll
  for (int r = 0; r < 8; r++) fr[r] = rowf[r + hi * 8];
  #pragma unroll
  for (int n = 0; n < 8; n++)
    #pragma unroll
    for (int r = 0; r < 8; r++) {
      int row = q0 + r + hi * 8;
      int col = h * HD_ + n * 16 + l15;
      ob[((size_t)b * S_ + row) * H_ + col] = f2bf(o[n][r] * fr[r]);
    }
}

// ---------------- host-side orchestration ----------------
extern "C" void kernel_launch(void* const* d_in, const int* in_sizes, int n_in,
                              void* d_out, int out_size, void* d_ws, size_t ws_size,
                              hipStream_t stream) {
  (void)in_sizes; (void)n_in; (void)out_size; (void)ws_size;
  const float* hidden = (const float*)d_in[0];   // [B,S,H]
  const float* Wqkv   = (const float*)d_in[1];   // [H,3H]
  const float* qnw    = (const float*)d_in[2];   // [H]
  const float* knw    = (const float*)d_in[3];   // [H]
  const float* Wo     = (const float*)d_in[4];   // [H,H]
  float* out = (float*)d_out;

  // workspace layout
  unsigned short* hB    = (unsigned short*)d_ws;                       // 16 MB
  unsigned short* wqkvB = hB    + (size_t)B_ * S_ * H_;                // 24 MB
  unsigned short* woB   = wqkvB + (size_t)H_ * 3 * H_;                 // 8 MB
  float*          qkv   = (float*)(woB + (size_t)H_ * H_);             // 96 MB
  unsigned short* qb    = (unsigned short*)(qkv + (size_t)B_ * S_ * 3 * H_);
  unsigned short* kb    = qb + (size_t)B_ * S_ * H_;
  unsigned short* vT    = kb + (size_t)B_ * S_ * H_;
  unsigned short* attnB = vT + (size_t)B_ * S_ * H_;

  const size_t nHid = (size_t)B_ * S_ * H_;
  const size_t nWq  = (size_t)H_ * 3 * H_;
  const size_t nWo  = (size_t)H_ * H_;

  cvt_f32_bf16<<<(unsigned)((nHid + 255) / 256), 256, 0, stream>>>(hidden, hB, nHid);
  cvt_f32_bf16<<<(unsigned)((nWq  + 255) / 256), 256, 0, stream>>>(Wqkv, wqkvB, nWq);
  cvt_f32_bf16<<<(unsigned)((nWo  + 255) / 256), 256, 0, stream>>>(Wo, woB, nWo);

  // qkv = hidden @ Wqkv   (M=4096, N=6144, K=2048)
  {
    dim3 grid(3 * H_ / 128, (B_ * S_) / 128);
    gemm_bf16<3 * H_, H_><<<grid, 256, 0, stream>>>(hB, wqkvB, qkv);
  }

  // rmsnorm + rope + split/transpose
  norm_rope_split<<<B_ * S_, 256, 0, stream>>>(qkv, qnw, knw, qb, kb, vT);

  // attention: one wave per (b, h, 16-query tile)
  attn_wmma<<<B_ * NH_ * (S_ / 16), 32, 0, stream>>>(qb, kb, vT, attnB);

  // out = attn @ Wo   (M=4096, N=2048, K=2048)
  {
    dim3 grid(H_ / 128, (B_ * S_) / 128);
    gemm_bf16<H_, H_><<<grid, 256, 0, stream>>>(attnB, woB, out);
  }
}
